// ImprovedLSTMDecoder_24154896073039
// MI455X (gfx1250) — matrix operand order repaired and checked by reading, not compiled, not run on hardware
//
#include <hip/hip_runtime.h>

// ---------------------------------------------------------------------------
// CDNA5 (gfx1250) attention-LSTM decoder. bf16 WMMA GEMMs (f32 accumulate),
// wave-per-16x16-tile mapping, weights pre-converted/transposed to bf16 once.
// ---------------------------------------------------------------------------

typedef __bf16 bf16;
typedef __attribute__((ext_vector_type(16))) __bf16 v16bf;
typedef __attribute__((ext_vector_type(8)))  float  v8f;

#define Bz 32
#define Sz 128
#define Rz 49
#define Ez 512
#define Hz 512
#define Az 512
#define Vz 10000
#define Tz 127          // S-1 decode steps
#define BR (Bz * Rz)    // 1568 attention rows

// ---------------- scalar conversion helpers (bit-exact RNE) ----------------
__device__ __forceinline__ bf16 f2bf(float f) {
  unsigned u = __builtin_bit_cast(unsigned, f);
  unsigned r = u + 0x7fffu + ((u >> 16) & 1u);
  unsigned short h = (unsigned short)(r >> 16);
  return __builtin_bit_cast(bf16, h);
}
__device__ __forceinline__ float bf2f(bf16 b) {
  unsigned short h = __builtin_bit_cast(unsigned short, b);
  unsigned u = ((unsigned)h) << 16;
  return __builtin_bit_cast(float, u);
}
__device__ __forceinline__ float sigm(float x) { return 1.f / (1.f + expf(-x)); }

// ---------------- WMMA fragment loaders (contiguous-k 16-bit order) ---------------
// A (MxK, row-major): lanes 0-15 rows M, K 0..15; lanes 16-31 K 16..31
__device__ __forceinline__ v16bf load_a(const bf16* A, int lda, int m0, int k0, int lane) {
  const bf16* p = A + (size_t)(m0 + (lane & 15)) * lda + (k0 + ((lane >> 4) << 4));
  v16bf a;
#pragma unroll
  for (int j = 0; j < 16; ++j) a[j] = p[j];
  return a;
}
// B (KxN) supplied as W[N,K] row-major (pre-transposed): lane holds one column,
// lanes 0-15 -> K 0..15, lanes 16-31 -> K 16..31. Contiguous 32B per lane.
__device__ __forceinline__ v16bf load_bt(const bf16* W, int ldw, int k0, int n0, int lane) {
  const bf16* p = W + (size_t)(n0 + (lane & 15)) * ldw + (k0 + ((lane >> 4) << 4));
  v16bf b;
#pragma unroll
  for (int j = 0; j < 16; ++j) b[j] = p[j];
  return b;
}

// ---------------- generic GEMM: C = act(A @ W^T + bias) --------------------
// wave-per-16x16 tile, 8 waves/block over consecutive N tiles. ACT: 0 none, 1 relu
template <int ACT>
__global__ __launch_bounds__(256) void gemm_bias_kernel(
    const bf16* __restrict__ A, int lda, const bf16* __restrict__ W, int ldw,
    const float* __restrict__ bias, float* __restrict__ C, int ldc, int N, int K) {
  int lane = threadIdx.x & 31, wave = threadIdx.x >> 5;
  int n0 = (blockIdx.x * 8 + wave) * 16;
  if (n0 >= N) return;
  int m0 = blockIdx.y * 16;
  v8f acc = {};
  for (int k0 = 0; k0 < K; k0 += 32) {
    if (k0 + 32 < K)
      __builtin_prefetch(W + (size_t)(n0 + (lane & 15)) * ldw + k0 + 32, 0, 1);
    v16bf a = load_a(A, lda, m0, k0, lane);
    v16bf b = load_bt(W, ldw, k0, n0, lane);
    acc = __builtin_amdgcn_wmma_f32_16x16x32_bf16(false, a, false, b, (short)0, acc,
                                                  false, false);
  }
  int col = n0 + (lane & 15);
  int rbase = m0 + ((lane >> 4) << 3);
  float bs = bias ? bias[col] : 0.f;
#pragma unroll
  for (int r = 0; r < 8; ++r) {
    float v = acc[r] + bs;
    if (ACT == 1) v = fmaxf(v, 0.f);
    C[(size_t)(rbase + r) * ldc + col] = v;
  }
}

// ------------- LSTM gate GEMM: C = A1@W1^T + A2@W2^T + b1 + b2 -------------
__global__ __launch_bounds__(256) void gemm2_gates_kernel(
    const bf16* __restrict__ A1, int lda1, const bf16* __restrict__ W1, int K1,
    const bf16* __restrict__ A2, int lda2, const bf16* __restrict__ W2, int K2,
    const float* __restrict__ b1, const float* __restrict__ b2,
    float* __restrict__ C, int ldc, int N) {
  int lane = threadIdx.x & 31, wave = threadIdx.x >> 5;
  int n0 = (blockIdx.x * 8 + wave) * 16;
  if (n0 >= N) return;
  int m0 = blockIdx.y * 16;
  v8f acc = {};
  for (int k0 = 0; k0 < K1; k0 += 32) {
    v16bf a = load_a(A1, lda1, m0, k0, lane);
    v16bf b = load_bt(W1, K1, k0, n0, lane);
    acc = __builtin_amdgcn_wmma_f32_16x16x32_bf16(false, a, false, b, (short)0, acc,
                                                  false, false);
  }
  for (int k0 = 0; k0 < K2; k0 += 32) {
    v16bf a = load_a(A2, lda2, m0, k0, lane);
    v16bf b = load_bt(W2, K2, k0, n0, lane);
    acc = __builtin_amdgcn_wmma_f32_16x16x32_bf16(false, a, false, b, (short)0, acc,
                                                  false, false);
  }
  int col = n0 + (lane & 15);
  int rbase = m0 + ((lane >> 4) << 3);
  float bs = b1[col] + b2[col];
#pragma unroll
  for (int r = 0; r < 8; ++r)
    C[(size_t)(rbase + r) * ldc + col] = acc[r] + bs;
}

// --- energy GEMM with fused final dot: e[m] += sum_n relu(tile+b1)*full2[n] ---
__global__ __launch_bounds__(256) void energy_kernel(
    const bf16* __restrict__ comb, const bf16* __restrict__ W1t,
    const float* __restrict__ b1, const float* __restrict__ full2,
    float* __restrict__ energy) {
  int lane = threadIdx.x & 31, wave = threadIdx.x >> 5;
  int n0 = (blockIdx.x * 8 + wave) * 16;   // N = 512 -> grid.x = 4
  int m0 = blockIdx.y * 16;                // M = 1568 -> grid.y = 98
  v8f acc = {};
  for (int k0 = 0; k0 < Az; k0 += 32) {
    v16bf a = load_a(comb, Az, m0, k0, lane);
    v16bf b = load_bt(W1t, Az, k0, n0, lane);
    acc = __builtin_amdgcn_wmma_f32_16x16x32_bf16(false, a, false, b, (short)0, acc,
                                                  false, false);
  }
  int col = n0 + (lane & 15);
  int rbase = m0 + ((lane >> 4) << 3);
  float bs = b1[col], f2 = full2[col];
#pragma unroll
  for (int r = 0; r < 8; ++r) {
    float s = fmaxf(acc[r] + bs, 0.f) * f2;
    s += __shfl_xor(s, 1, 32);
    s += __shfl_xor(s, 2, 32);
    s += __shfl_xor(s, 4, 32);
    s += __shfl_xor(s, 8, 32);   // sum over the 16 lanes sharing row m
    if ((lane & 15) == 0) atomicAdd(&energy[rbase + r], s);
  }
}

// ---------------- one-time weight conversion kernels -----------------------
__global__ void cvt_kernel(const float* __restrict__ src, bf16* __restrict__ dst, int n) {
  int i = blockIdx.x * 256 + threadIdx.x;
  if (i < n) dst[i] = f2bf(src[i]);
}
// src is [K,N] row-major, dst becomes [N,K] row-major (so GEMM B loads are dense)
__global__ void cvt_tr_kernel(const float* __restrict__ src, bf16* __restrict__ dst,
                              int K, int N) {
  int i = blockIdx.x * 256 + threadIdx.x;
  if (i < K * N) {
    int k = i / N, n = i - k * N;
    dst[(size_t)n * K + k] = f2bf(src[i]);
  }
}

// -------- initial state: mean over regions -> tanh(Linear) for h0/c0 -------
__global__ __launch_bounds__(256) void init_state_kernel(
    const float* __restrict__ features,
    const float* __restrict__ ihW, const float* __restrict__ ihb,
    const float* __restrict__ icW, const float* __restrict__ icb,
    float* __restrict__ c0, float* __restrict__ c1,
    bf16* __restrict__ hb0, bf16* __restrict__ hb1) {
  __shared__ float smean[Ez];
  int b = blockIdx.x, tid = threadIdx.x;
  for (int e = tid; e < Ez; e += 256) {
    float s = 0.f;
    const float* fp = features + (size_t)b * Rz * Ez + e;
    for (int r = 0; r < Rz; ++r) s += fp[(size_t)r * Ez];
    smean[e] = s * (1.f / Rz);
  }
  __syncthreads();
  for (int j = tid; j < Hz; j += 256) {
    float sh = ihb[j], sc = icb[j];
    for (int k = 0; k < Ez; ++k) {
      float m = smean[k];
      sh += m * ihW[(size_t)k * Hz + j];
      sc += m * icW[(size_t)k * Hz + j];
    }
    float h = tanhf(sh), c = tanhf(sc);
    c0[b * Hz + j] = c; c1[b * Hz + j] = c;
    hb0[b * Hz + j] = f2bf(h); hb1[b * Hz + j] = f2bf(h);
  }
}

// ------- comb = bf16(relu(att1 + att2 broadcast)); also seed energy --------
__global__ void comb_kernel(const float* __restrict__ att1, const float* __restrict__ att2,
                            const float* __restrict__ full2_b, bf16* __restrict__ comb,
                            float* __restrict__ energy) {
  int idx = blockIdx.x * 256 + threadIdx.x;         // n = 1568*512 exactly
  int m = idx >> 9, e = idx & 511, b = m / Rz;
  float v = att1[idx] + att2[b * Az + e];
  comb[idx] = f2bf(fmaxf(v, 0.f));
  if (idx < BR) energy[idx] = full2_b[0];
}

// ---- softmax over regions, context, context gate, build x / combo rows ----
__global__ __launch_bounds__(256) void attn_ctx_gate_kernel(
    const float* __restrict__ energy, const float* __restrict__ features,
    const bf16* __restrict__ hb1,
    const float* __restrict__ g1W, const float* __restrict__ g1b,
    const float* __restrict__ g2W, const float* __restrict__ g2b,
    const float* __restrict__ embW, const int* __restrict__ captions, int t,
    bf16* __restrict__ x_bf, bf16* __restrict__ combo_bf,
    float* __restrict__ alphas_out) {
  __shared__ float sA[64];
  __shared__ float sCtx[Ez];
  __shared__ float sred[256];
  __shared__ float sGate;
  int b = blockIdx.x, tid = threadIdx.x;
  if (tid < Rz) sA[tid] = energy[b * Rz + tid];
  __syncthreads();
  if (tid == 0) {
    float m = sA[0];
    for (int r = 1; r < Rz; ++r) m = fmaxf(m, sA[r]);
    float s = 0.f;
    for (int r = 0; r < Rz; ++r) { sA[r] = expf(sA[r] - m); s += sA[r]; }
    float inv = 1.f / s;
    for (int r = 0; r < Rz; ++r) sA[r] *= inv;
  }
  __syncthreads();
  if (tid < Rz) alphas_out[((size_t)b * Tz + t) * Rz + tid] = sA[tid];
  for (int e = tid; e < Ez; e += 256) {
    float s = 0.f;
    const float* fp = features + (size_t)b * Rz * Ez + e;
    for (int r = 0; r < Rz; ++r) s += sA[r] * fp[(size_t)r * Ez];
    sCtx[e] = s;
  }
  __syncthreads();
  // context gate MLP: 1024 -> 256 (relu) -> 1 (sigmoid)
  int j = tid;
  float d = g1b[j];
  for (int k = 0; k < Ez; ++k) d += sCtx[k] * g1W[(size_t)k * 256 + j];
  for (int k = 0; k < Hz; ++k) d += bf2f(hb1[b * Hz + k]) * g1W[(size_t)(Ez + k) * 256 + j];
  sred[tid] = fmaxf(d, 0.f) * g2W[j];
  __syncthreads();
  for (int s = 128; s; s >>= 1) {
    if (tid < s) sred[tid] += sred[tid + s];
    __syncthreads();
  }
  if (tid == 0) sGate = sigm(sred[0] + g2b[0]);
  __syncthreads();
  float gate = sGate;
  int cap = captions[b * Sz + t];
  for (int e = tid; e < Ez; e += 256) {
    float cv = sCtx[e] * gate;
    x_bf[b * 2 * Ez + Ez + e] = f2bf(cv);       // x = [emb_t, context]
    combo_bf[b * 2 * Hz + Hz + e] = f2bf(cv);   // combo = [h1n, context]
    x_bf[b * 2 * Ez + e] = f2bf(embW[(size_t)cap * Ez + e]);
  }
}

// ------------------------------ LSTM cell ----------------------------------
__global__ void lstm_cell_kernel(const float* __restrict__ gates, float* __restrict__ c,
                                 bf16* __restrict__ h_bf, bf16* __restrict__ h_bf2,
                                 int h2_ld, int n) {
  int idx = blockIdx.x * 256 + threadIdx.x;
  if (idx >= n) return;
  int b = idx >> 9, j = idx & 511;
  const float* g = gates + (size_t)b * 4 * Hz;
  float ig = sigm(g[j]), fg = sigm(g[Hz + j]);
  float gg = tanhf(g[2 * Hz + j]), og = sigm(g[3 * Hz + j]);
  float c2 = fg * c[idx] + ig * gg;
  float h2 = og * tanhf(c2);
  c[idx] = c2;
  h_bf[idx] = f2bf(h2);
  if (h_bf2) h_bf2[(size_t)b * h2_ld + j] = f2bf(h2);
}

// ------------------------- LayerNorm + ReLU (per row) ----------------------
__global__ __launch_bounds__(256) void ln_relu_kernel(
    const float* __restrict__ z, const float* __restrict__ g,
    const float* __restrict__ be, bf16* __restrict__ zr) {
  __shared__ float red[256];
  int b = blockIdx.x, tid = threadIdx.x;
  float v0 = z[b * Hz + tid], v1 = z[b * Hz + 256 + tid];
  red[tid] = v0 + v1;
  __syncthreads();
  for (int s = 128; s; s >>= 1) { if (tid < s) red[tid] += red[tid + s]; __syncthreads(); }
  float mu = red[0] * (1.f / Hz);
  __syncthreads();
  float d0 = v0 - mu, d1 = v1 - mu;
  red[tid] = d0 * d0 + d1 * d1;
  __syncthreads();
  for (int s = 128; s; s >>= 1) { if (tid < s) red[tid] += red[tid + s]; __syncthreads(); }
  float rstd = rsqrtf(red[0] * (1.f / Hz) + 1e-5f);
  zr[b * Hz + tid]       = f2bf(fmaxf(d0 * rstd * g[tid] + be[tid], 0.f));
  zr[b * Hz + 256 + tid] = f2bf(fmaxf(d1 * rstd * g[256 + tid] + be[256 + tid], 0.f));
}

// ===========================================================================
extern "C" void kernel_launch(void* const* d_in, const int* in_sizes, int n_in,
                              void* d_out, int out_size, void* d_ws, size_t ws_size,
                              hipStream_t stream) {
  (void)in_sizes; (void)n_in; (void)out_size; (void)ws_size;

  // ---- inputs (setup_inputs order) ----
  const float* features = (const float*)d_in[0];
  const int*   captions = (const int*)d_in[1];
  const float* emb_W    = (const float*)d_in[2];
  const float* enc_W    = (const float*)d_in[3];
  const float* enc_b    = (const float*)d_in[4];
  const float* dec_W    = (const float*)d_in[5];
  const float* dec_b    = (const float*)d_in[6];
  const float* full1_W  = (const float*)d_in[7];
  const float* full1_b  = (const float*)d_in[8];
  const float* full2_W  = (const float*)d_in[9];
  const float* full2_b  = (const float*)d_in[10];
  const float* gate1_W  = (const float*)d_in[11];
  const float* gate1_b  = (const float*)d_in[12];
  const float* gate2_W  = (const float*)d_in[13];
  const float* gate2_b  = (const float*)d_in[14];
  const float* W_ih0    = (const float*)d_in[15];
  const float* b_ih0    = (const float*)d_in[16];
  const float* W_hh0    = (const float*)d_in[17];
  const float* b_hh0    = (const float*)d_in[18];
  const float* W_ih1    = (const float*)d_in[19];
  const float* b_ih1    = (const float*)d_in[20];
  const float* W_hh1    = (const float*)d_in[21];
  const float* b_hh1    = (const float*)d_in[22];
  const float* ih_W     = (const float*)d_in[23];
  const float* ih_b     = (const float*)d_in[24];
  const float* ic_W     = (const float*)d_in[25];
  const float* ic_b     = (const float*)d_in[26];
  const float* fc1_W    = (const float*)d_in[27];
  const float* fc1_b    = (const float*)d_in[28];
  const float* ln_g     = (const float*)d_in[29];
  const float* ln_b     = (const float*)d_in[30];
  const float* fc2_W    = (const float*)d_in[31];
  const float* fc2_b    = (const float*)d_in[32];

  float* logits = (float*)d_out;                                  // [B,127,V]
  float* alphas = logits + (size_t)Bz * Tz * Vz;                  // [B,127,R]

  // ---- workspace carve-up ----
  char* w = (char*)d_ws;
  size_t off = 0;
  auto alloc = [&](size_t bytes) {
    void* p = w + off;
    off += (bytes + 255) & ~(size_t)255;
    return p;
  };
  bf16* wtb_enc   = (bf16*)alloc(sizeof(bf16) * Ez * Az);
  bf16* wtb_dec   = (bf16*)alloc(sizeof(bf16) * Hz * Az);
  bf16* wtb_full1 = (bf16*)alloc(sizeof(bf16) * Az * Az);
  bf16* wb_ih0    = (bf16*)alloc(sizeof(bf16) * 4 * Hz * 2 * Ez);
  bf16* wb_hh0    = (bf16*)alloc(sizeof(bf16) * 4 * Hz * Hz);
  bf16* wb_ih1    = (bf16*)alloc(sizeof(bf16) * 4 * Hz * Hz);
  bf16* wb_hh1    = (bf16*)alloc(sizeof(bf16) * 4 * Hz * Hz);
  bf16* wtb_fc1   = (bf16*)alloc(sizeof(bf16) * (Hz + Ez) * Hz);
  bf16* wtb_fc2   = (bf16*)alloc(sizeof(bf16) * Hz * Vz);
  bf16* feat_bf   = (bf16*)alloc(sizeof(bf16) * BR * Ez);
  bf16* comb_bf   = (bf16*)alloc(sizeof(bf16) * BR * Az);
  bf16* x_bf      = (bf16*)alloc(sizeof(bf16) * Bz * 2 * Ez);
  bf16* combo_bf  = (bf16*)alloc(sizeof(bf16) * Bz * 2 * Hz);
  bf16* hb0       = (bf16*)alloc(sizeof(bf16) * Bz * Hz);
  bf16* hb1       = (bf16*)alloc(sizeof(bf16) * Bz * Hz);
  bf16* zr_bf     = (bf16*)alloc(sizeof(bf16) * Bz * Hz);
  float* att1     = (float*)alloc(sizeof(float) * BR * Az);
  float* att2     = (float*)alloc(sizeof(float) * Bz * Az);
  float* energy   = (float*)alloc(sizeof(float) * BR);
  float* c0       = (float*)alloc(sizeof(float) * Bz * Hz);
  float* c1       = (float*)alloc(sizeof(float) * Bz * Hz);
  float* gates    = (float*)alloc(sizeof(float) * Bz * 4 * Hz);
  float* z        = (float*)alloc(sizeof(float) * Bz * Hz);

  auto cvt = [&](const float* s, bf16* d, int n) {
    cvt_kernel<<<(n + 255) / 256, 256, 0, stream>>>(s, d, n);
  };
  auto cvt_tr = [&](const float* s, bf16* d, int K, int N) {
    cvt_tr_kernel<<<(K * N + 255) / 256, 256, 0, stream>>>(s, d, K, N);
  };

  // ---- one-time prep: bf16 conversion (+transpose of [K,N] weights) ----
  cvt(W_ih0, wb_ih0, 4 * Hz * 2 * Ez);
  cvt(W_hh0, wb_hh0, 4 * Hz * Hz);
  cvt(W_ih1, wb_ih1, 4 * Hz * Hz);
  cvt(W_hh1, wb_hh1, 4 * Hz * Hz);
  cvt(features, feat_bf, BR * Ez);
  cvt_tr(enc_W, wtb_enc, Ez, Az);
  cvt_tr(dec_W, wtb_dec, Hz, Az);
  cvt_tr(full1_W, wtb_full1, Az, Az);
  cvt_tr(fc1_W, wtb_fc1, Hz + Ez, Hz);
  cvt_tr(fc2_W, wtb_fc2, Hz, Vz);

  init_state_kernel<<<Bz, 256, 0, stream>>>(features, ih_W, ih_b, ic_W, ic_b,
                                            c0, c1, hb0, hb1);

  // att1 = features @ enc_att_W + b   [1568,512], loop-invariant
  gemm_bias_kernel<0><<<dim3(4, BR / 16), 256, 0, stream>>>(
      feat_bf, Ez, wtb_enc, Ez, enc_b, att1, Az, Az, Ez);

  // ---- sequential decode ----
  for (int t = 0; t < Tz; ++t) {
    // att2 = h_top @ dec_att_W + b    [32,512]
    gemm_bias_kernel<0><<<dim3(4, 2), 256, 0, stream>>>(
        hb1, Hz, wtb_dec, Hz, dec_b, att2, Az, Az, Hz);
    // comb = relu(att1 + att2), seed energy with full2_b
    comb_kernel<<<(BR * Az) / 256, 256, 0, stream>>>(att1, att2, full2_b, comb_bf, energy);
    // energy += relu(comb @ full1_W + b1) . full2
    energy_kernel<<<dim3(4, BR / 16), 256, 0, stream>>>(comb_bf, wtb_full1, full1_b,
                                                        full2_W, energy);
    // softmax, context, gate, build x/combo rows, emit alphas
    attn_ctx_gate_kernel<<<Bz, 256, 0, stream>>>(energy, features, hb1,
                                                 gate1_W, gate1_b, gate2_W, gate2_b,
                                                 emb_W, captions, t, x_bf, combo_bf, alphas);
    // layer-0 gates + cell
    gemm2_gates_kernel<<<dim3(16, 2), 256, 0, stream>>>(
        x_bf, 2 * Ez, wb_ih0, 2 * Ez, hb0, Hz, wb_hh0, Hz,
        b_ih0, b_hh0, gates, 4 * Hz, 4 * Hz);
    lstm_cell_kernel<<<(Bz * Hz) / 256, 256, 0, stream>>>(gates, c0, hb0, nullptr, 0,
                                                          Bz * Hz);
    // layer-1 gates + cell (writes h1n into combo front half)
    gemm2_gates_kernel<<<dim3(16, 2), 256, 0, stream>>>(
        hb0, Hz, wb_ih1, Hz, hb1, Hz, wb_hh1, Hz,
        b_ih1, b_hh1, gates, 4 * Hz, 4 * Hz);
    lstm_cell_kernel<<<(Bz * Hz) / 256, 256, 0, stream>>>(gates, c1, hb1, combo_bf,
                                                          2 * Hz, Bz * Hz);
    // output head
    gemm_bias_kernel<0><<<dim3(4, 2), 256, 0, stream>>>(
        combo_bf, 2 * Hz, wtb_fc1, Hz + Ez, fc1_b, z, Hz, Hz, Hz + Ez);
    ln_relu_kernel<<<Bz, 256, 0, stream>>>(z, ln_g, ln_b, zr_bf);
    gemm_bias_kernel<0><<<dim3(79, 2), 256, 0, stream>>>(
        zr_bf, Hz, wtb_fc2, Hz, fc2_b, logits + (size_t)t * Vz, Tz * Vz, Vz, Hz);
  }
}
